// GroupedQueryAttention_68178310857163
// MI455X (gfx1250) — matrix operand order repaired and checked by reading, not compiled, not run on hardware
//
#include <hip/hip_runtime.h>

// ---------------------------------------------------------------------------
// GQA for MI455X (gfx1250, wave32, WMMA).
// All matmuls via v_wmma_f32_16x16x32_f16 (f16 in, f32 accum).
// Problem: B=2 N=2048 E=2048 H=16 D=128 HKV=4 G=4 KVD=512.
// ~155 GFLOP vs ~110MB traffic -> heavily compute bound -> WMMA everywhere.
// V is produced TRANSPOSED by its projection GEMM so the P*V B-fragments are
// contiguous b128 loads. Attention blocks share K/V tiles across 4 waves via
// LDS, staged with GLOBAL_LOAD_ASYNC_TO_LDS_B128 when available.
// ---------------------------------------------------------------------------

typedef _Float16 half_t;
typedef __attribute__((ext_vector_type(16))) _Float16 v16h;
typedef __attribute__((ext_vector_type(8)))  _Float16 v8h;
typedef __attribute__((ext_vector_type(8)))  float    v8f;
typedef __attribute__((__vector_size__(4 * sizeof(int)))) int b128v;

constexpr int Bc  = 2;
constexpr int Nc  = 2048;
constexpr int Ec  = 2048;
constexpr int Hc  = 16;
constexpr int Gc  = 4;
constexpr int Dc  = 128;    // q head dim
constexpr int KVD = 512;    // kv "head-concat" dim (HKV * D)

// ---------------- async global->LDS copy (CDNA5 path with fallback) ---------
#if __has_builtin(__builtin_amdgcn_global_load_async_to_lds_b128)
#define HAVE_ASYNC_COPY 1
#endif

__device__ __forceinline__ void copy16_g2l(const half_t* g, half_t* l) {
#ifdef HAVE_ASYNC_COPY
  __builtin_amdgcn_global_load_async_to_lds_b128(
      (__attribute__((address_space(1))) b128v*)g,
      (__attribute__((address_space(3))) b128v*)l, 0, 0);
#else
  *(v8h*)l = *(const v8h*)g;
#endif
}

__device__ __forceinline__ void wait_async_copies() {
#ifdef HAVE_ASYNC_COPY
#if __has_builtin(__builtin_amdgcn_s_wait_asynccnt)
  __builtin_amdgcn_s_wait_asynccnt(0);
#else
  asm volatile("s_wait_asynccnt 0" ::: "memory");
#endif
#endif
}

// ---------------- fragment helpers -----------------------------------------
// A-frag (16x32 f16, M x K): lane L holds row (L&15); ISA layout:
//   halves 0..7  -> K = kb..kb+7      (kb = (L>=16)?8:0)
//   halves 8..15 -> K = 16+kb..16+kb+7
__device__ __forceinline__ v16h load_a_rm(const half_t* base, int ld) {
  const int lane = threadIdx.x & 31;
  const int row  = lane & 15;
  const int kb   = (lane >> 4) << 3;
  const half_t* p = base + (size_t)row * ld + kb;
  v8h lo = *(const v8h*)(p);
  v8h hi = *(const v8h*)(p + 16);
  v16h r;
#pragma unroll
  for (int j = 0; j < 8; ++j) { r[j] = lo[j]; r[j + 8] = hi[j]; }
  return r;
}

// B-frag (32x16 f16, K x N) where column n of B is a contiguous row of
// length>=32 in memory (B = W^T with W row-major (Out,In), or K/V^T tiles).
// Lane L holds column (L&15); halves 0..7 -> K = kb..kb+7, 8..15 -> K = 16+kb..16+kb+7 (kb = (L>=16)?8:0).
__device__ __forceinline__ v16h load_b_rows(const half_t* base, int ld) {
  const int lane = threadIdx.x & 31;
  const int col  = lane & 15;
  const int kb   = (lane >> 4) << 3;
  const half_t* p = base + (size_t)col * ld + kb;
  v8h lo = *(const v8h*)(p);
  v8h hi = *(const v8h*)(p + 16);
  v16h r;
#pragma unroll
  for (int j = 0; j < 8; ++j) { r[j] = lo[j]; r[j + 8] = hi[j]; }
  return r;
}

__device__ __forceinline__ v8f wmma32(v16h a, v16h b, v8f c) {
  return __builtin_amdgcn_wmma_f32_16x16x32_f16(false, a, false, b,
                                                (short)0, c, false, false);
}

// ---------------- f32 -> f16 conversion -------------------------------------
__global__ void f32_to_f16(const float* __restrict__ in,
                           half_t* __restrict__ out, int n) {
  int i = blockIdx.x * blockDim.x + threadIdx.x;
  if (i < n) out[i] = (half_t)in[i];
}

// ---------------- GEMM: Y(M,Nout) = X(M,K) @ W(Nout,K)^T --------------------
// One 16x16 output tile per wave, full-K loop, f32 accum.
template <typename OutT>
__global__ void __launch_bounds__(128)
gemm_xwt(const half_t* __restrict__ X, const half_t* __restrict__ W,
         OutT* __restrict__ Y, int M, int Nout, int K) {
  const int wave = blockIdx.x * (blockDim.x >> 5) + (threadIdx.x >> 5);
  const int ntn  = Nout >> 4;
  const int rt   = wave / ntn;
  const int ct   = wave % ntn;
  if (rt * 16 >= M) return;

  const half_t* xb = X + (size_t)rt * 16 * K;
  const half_t* wb = W + (size_t)ct * 16 * K;

  v8f acc = {};
  for (int kk = 0; kk < K; kk += 32) {
    if (kk + 512 < K) {
      __builtin_prefetch(xb + kk + 512, 0, 3);
      __builtin_prefetch(wb + kk + 512, 0, 3);
    }
    v16h a = load_a_rm(xb + kk, K);
    v16h b = load_b_rows(wb + kk, K);
    acc = wmma32(a, b, acc);
  }

  const int lane = threadIdx.x & 31;
  const int m0   = (lane >> 4) << 3;
  const int n    = lane & 15;
  OutT* yb = Y + ((size_t)rt * 16 + m0) * Nout + ct * 16 + n;
#pragma unroll
  for (int r = 0; r < 8; ++r) yb[(size_t)r * Nout] = (OutT)acc[r];
}

// ---- GEMM with transposed store: Yt(batch, Nout, seqN), batch = M/seqN -----
// Used for V so that attention's P*V B-frags read contiguous V^T rows.
__global__ void __launch_bounds__(128)
gemm_xwt_tout(const half_t* __restrict__ X, const half_t* __restrict__ W,
              half_t* __restrict__ Yt, int M, int Nout, int K, int seqN) {
  const int wave = blockIdx.x * (blockDim.x >> 5) + (threadIdx.x >> 5);
  const int ntn  = Nout >> 4;
  const int rt   = wave / ntn;
  const int ct   = wave % ntn;
  if (rt * 16 >= M) return;

  const half_t* xb = X + (size_t)rt * 16 * K;
  const half_t* wb = W + (size_t)ct * 16 * K;

  v8f acc = {};
  for (int kk = 0; kk < K; kk += 32) {
    v16h a = load_a_rm(xb + kk, K);
    v16h b = load_b_rows(wb + kk, K);
    acc = wmma32(a, b, acc);
  }

  const int lane = threadIdx.x & 31;
  const int m0   = (lane >> 4) << 3;   // row (token) group within tile
  const int n    = lane & 15;          // output channel within tile
  const int mg   = rt * 16;
  const int bb   = mg / seqN;          // batch (uniform: 16 | seqN)
  const int s    = mg - bb * seqN + m0;
  half_t* yb = Yt + ((size_t)bb * Nout + ct * 16 + n) * seqN + s;
  v8h ov;
#pragma unroll
  for (int r = 0; r < 8; ++r) ov[r] = (half_t)acc[r];
  *(v8h*)yb = ov;                      // 8 consecutive tokens of one channel
}

// ---------------- flash attention -------------------------------------------
// Block = 4 waves on the SAME (b, head), 4 consecutive 16-row query tiles.
// Per 64-key step: K tile (64x128) and V^T tile (128x64) staged to LDS once
// per block (async copies), consumed by all 4 waves as ds_load_b128 B-frags.
__global__ void __launch_bounds__(128)
gqa_attn(const half_t* __restrict__ Q, const half_t* __restrict__ Kh,
         const half_t* __restrict__ Vt, half_t* __restrict__ O) {
  __shared__ half_t ldsK[64 * 128];       // 16 KB: keys s0..s0+63, d=0..127
  __shared__ half_t ldsV[128 * 64];       // 16 KB: V^T, d rows x 64 keys
  __shared__ half_t ldsP[4][16 * 64];     // 2 KB per wave: P re-layout

  const int wavein = threadIdx.x >> 5;
  const int qt     = (blockIdx.x & 31) * 4 + wavein;  // query tile
  const int head   = (blockIdx.x >> 5) & (Hc - 1);
  const int b      = blockIdx.x >> 9;                 // /(32*16)
  const int hkv    = head / Gc;
  const int lane   = threadIdx.x & 31;
  const int m0     = (lane >> 4) << 3;    // C/D-layout row group
  const int nlane  = lane & 15;           // C/D-layout column

  const half_t* qbase  = Q  + ((size_t)(b * Nc + qt * 16)) * Ec + head * Dc;
  const half_t* kbase  = Kh + ((size_t)(b * Nc)) * KVD + hkv * Dc;
  const half_t* vtbase = Vt + ((size_t)b * KVD + hkv * Dc) * Nc; // (d, s) ld=Nc
  half_t* pl = &ldsP[wavein][0];

  // Q tile resident: 16x128 = 4 A-frags of K=32
  v16h qf[4];
#pragma unroll
  for (int i = 0; i < 4; ++i) qf[i] = load_a_rm(qbase + 32 * i, Ec);

  v8f o[8];                               // 16x128 f32 accumulator
#pragma unroll
  for (int cb = 0; cb < 8; ++cb) o[cb] = {};
  float mrow[8], lrow[8];
#pragma unroll
  for (int r = 0; r < 8; ++r) { mrow[r] = -3.0e38f; lrow[r] = 0.f; }

  const float scale = 0.08838834764831845f; // 1/sqrt(128)
  const int   t     = threadIdx.x;

  for (int s0 = 0; s0 < Nc; s0 += 64) {
    // ---- cooperative staging: 2 x 16KB, 128 threads x 16 x b128 copies ----
    const half_t* kTile = kbase + (size_t)s0 * KVD;   // 64 rows x 128 halves
    const half_t* vTile = vtbase + s0;                // 128 rows x 64 halves
#pragma unroll
    for (int i = 0; i < 8; ++i) {
      const int c   = t + i * 128;      // chunk 0..1023 (16B chunks)
      const int off = c * 8;            // in halves
      const int kr  = off >> 7, kc = off & 127;
      copy16_g2l(kTile + (size_t)kr * KVD + kc, &ldsK[off]);
      const int vr  = off >> 6, vc = off & 63;
      copy16_g2l(vTile + (size_t)vr * Nc + vc, &ldsV[off]);
    }
    wait_async_copies();
    __syncthreads();

    // ---- S = Q K^T for a 16x64 score tile (16 WMMAs, B-frags from LDS) ----
    v8f sf[4];
#pragma unroll
    for (int t4 = 0; t4 < 4; ++t4) {
      sf[t4] = {};
#pragma unroll
      for (int i = 0; i < 4; ++i) {
        v16h kf = load_b_rows(&ldsK[(t4 * 16) * 128 + 32 * i], 128);
        sf[t4] = wmma32(qf[i], kf, sf[t4]);
      }
    }

    // ---- online softmax over the 64 new columns ----
    float alpha[8];
#pragma unroll
    for (int r = 0; r < 8; ++r) {
      float m = -3.0e38f;
#pragma unroll
      for (int t4 = 0; t4 < 4; ++t4) {
        float v = sf[t4][r] * scale;
        sf[t4][r] = v;
        m = fmaxf(m, v);
      }
      m = fmaxf(m, __shfl_xor(m, 1, 32));
      m = fmaxf(m, __shfl_xor(m, 2, 32));
      m = fmaxf(m, __shfl_xor(m, 4, 32));
      m = fmaxf(m, __shfl_xor(m, 8, 32));
      float mnew = fmaxf(mrow[r], m);
      alpha[r] = __expf(mrow[r] - mnew);
      mrow[r]  = mnew;
      float rs = 0.f;
#pragma unroll
      for (int t4 = 0; t4 < 4; ++t4) {
        float p = __expf(sf[t4][r] - mnew);
        rs += p;
        pl[(m0 + r) * 64 + t4 * 16 + nlane] = (half_t)p;   // stage P
      }
      rs += __shfl_xor(rs, 1, 32);
      rs += __shfl_xor(rs, 2, 32);
      rs += __shfl_xor(rs, 4, 32);
      rs += __shfl_xor(rs, 8, 32);
      lrow[r] = alpha[r] * lrow[r] + rs;
    }
#pragma unroll
    for (int cb = 0; cb < 8; ++cb)
#pragma unroll
      for (int r = 0; r < 8; ++r) o[cb][r] *= alpha[r];

    // make staged P visible before re-reading in A-frag layout (same wave)
    asm volatile("s_wait_dscnt 0" ::: "memory");

    // ---- O += P V : K=64 contraction, V^T B-frags straight from LDS -------
    v16h pa0 = load_a_rm(pl, 64);        // s =  0..31
    v16h pa1 = load_a_rm(pl + 32, 64);   // s = 32..63
#pragma unroll
    for (int cb = 0; cb < 8; ++cb) {
      v16h v0 = load_b_rows(&ldsV[(cb * 16) * 64 + 0], 64);
      v16h v1 = load_b_rows(&ldsV[(cb * 16) * 64 + 32], 64);
      o[cb] = wmma32(pa0, v0, o[cb]);
      o[cb] = wmma32(pa1, v1, o[cb]);
    }
    __syncthreads();                     // staged tiles fully consumed
  }

  // ---- normalize + store (b, n, head*D + d) ----
#pragma unroll
  for (int r = 0; r < 8; ++r) {
    float inv = 1.0f / lrow[r];
    size_t rowoff = ((size_t)(b * Nc + qt * 16 + m0 + r)) * Ec + head * Dc;
#pragma unroll
    for (int cb = 0; cb < 8; ++cb)
      O[rowoff + cb * 16 + nlane] = (half_t)(o[cb][r] * inv);
  }
}

// ---------------------------------------------------------------------------
extern "C" void kernel_launch(void* const* d_in, const int* in_sizes, int n_in,
                              void* d_out, int out_size, void* d_ws,
                              size_t ws_size, hipStream_t stream) {
  const float* x  = (const float*)d_in[0];
  const float* Wq = (const float*)d_in[1];
  const float* Wk = (const float*)d_in[2];
  const float* Wv = (const float*)d_in[3];
  const float* Wo = (const float*)d_in[4];
  float* out = (float*)d_out;

  const int BNE = Bc * Nc * Ec;        // 8388608
  const int EE  = Ec * Ec;             // 4194304
  const int KE  = KVD * Ec;            // 1048576
  const int BNK = Bc * Nc * KVD;       // 2097152

  half_t* w   = (half_t*)d_ws;
  half_t* Xh  = w;            w += BNE;
  half_t* Wqh = w;            w += EE;
  half_t* Wkh = w;            w += KE;
  half_t* Wvh = w;            w += KE;
  half_t* Woh = w;            w += EE;
  half_t* Qh  = w;            w += BNE;
  half_t* Khd = w;            w += BNK;
  half_t* Vth = w;            w += BNK;   // V^T: (B, KVD, N)
  half_t* Oh  = w;            w += BNE;

  // 1) downconvert activations + weights to f16
  f32_to_f16<<<(BNE + 255) / 256, 256, 0, stream>>>(x, Xh, BNE);
  f32_to_f16<<<(EE  + 255) / 256, 256, 0, stream>>>(Wq, Wqh, EE);
  f32_to_f16<<<(KE  + 255) / 256, 256, 0, stream>>>(Wk, Wkh, KE);
  f32_to_f16<<<(KE  + 255) / 256, 256, 0, stream>>>(Wv, Wvh, KE);
  f32_to_f16<<<(EE  + 255) / 256, 256, 0, stream>>>(Wo, Woh, EE);

  const int M = Bc * Nc;               // 4096 token rows

  // 2) projections: Q = X Wq^T, K = X Wk^T, V^T = (X Wv^T)^T
  {
    int tiles = (M / 16) * (Ec / 16);  // 32768
    gemm_xwt<half_t><<<tiles / 4, 128, 0, stream>>>(Xh, Wqh, Qh, M, Ec, Ec);
  }
  {
    int tiles = (M / 16) * (KVD / 16); // 8192
    gemm_xwt<half_t><<<tiles / 4, 128, 0, stream>>>(Xh, Wkh, Khd, M, KVD, Ec);
    gemm_xwt_tout<<<tiles / 4, 128, 0, stream>>>(Xh, Wvh, Vth, M, KVD, Ec, Nc);
  }

  // 3) grouped-query flash attention (block = 4 q-tiles of one head)
  {
    int blocks = Bc * Hc * (Nc / 16 / 4);  // 1024
    gqa_attn<<<blocks, 128, 0, stream>>>(Qh, Khd, Vth, Oh);
  }

  // 4) output projection (f32 result straight to d_out)
  {
    int tiles = (M / 16) * (Ec / 16);
    gemm_xwt<float><<<tiles / 4, 128, 0, stream>>>(Oh, Woh, out, M, Ec, Ec);
  }
}
